// KGramMLPSeqModel_45165876085584
// MI455X (gfx1250) — compile-verified, matrix operand order checked by reading, not run on hardware
//
#include <hip/hip_runtime.h>
#include <hip/hip_bf16.h>

// ---------------------------------------------------------------------------
// K-gram MLP seq model for gfx1250 (MI455X).
//   - bf16 WMMA (v_wmma_f32_16x16x32_bf16) for both GEMMs, fp32 accumulate
//   - async global->LDS double-buffered pipeline (ASYNCcnt / s_wait_asynccnt)
//   - LDS-tiled transpose for weight downconversion (coalesced both ways)
// ---------------------------------------------------------------------------

typedef __attribute__((ext_vector_type(16))) __bf16        v16bf;
typedef __attribute__((ext_vector_type(8)))  float         v8f;
typedef __attribute__((ext_vector_type(4)))  unsigned int  v4u;
typedef __attribute__((ext_vector_type(4)))  __bf16        v4bf;

union FragBF16 {          // one WMMA 16-bit operand fragment: 8 VGPRs
    v16bf v;
    v4u   q[2];           // two 16-byte loads
};

static constexpr int KCTX = 3;
static constexpr int T_   = 1024;
static constexpr int B_   = 4;
static constexpr int V_   = 32000;
static constexpr int E_   = 1024;
static constexpr int M_   = T_ * B_;      // 4096 GEMM rows
static constexpr int K1_  = KCTX * E_;    // 3072
static constexpr int H_   = 2 * E_;       // 2048

// --------------------------- gfx1250 async helpers -------------------------

__device__ __forceinline__ void async_copy16(void* lds_dst, const void* gsrc)
{
    // VDST = LDS byte address (wave-relative; low 32 bits of the generic
    // pointer), VADDR = 64-bit global address, saddr = off.
    unsigned l = (unsigned)(unsigned long long)lds_dst;
    asm volatile("global_load_async_to_lds_b128 %0, %1, off"
                 :: "v"(l), "v"(gsrc)
                 : "memory");
}

__device__ __forceinline__ void wait_async_le5()
{
#if __has_builtin(__builtin_amdgcn_s_wait_asynccnt)
    __builtin_amdgcn_s_wait_asynccnt(5);
#else
    asm volatile("s_wait_asynccnt 5" ::: "memory");
#endif
}

__device__ __forceinline__ void wait_async_le0()
{
#if __has_builtin(__builtin_amdgcn_s_wait_asynccnt)
    __builtin_amdgcn_s_wait_asynccnt(0);
#else
    asm volatile("s_wait_asynccnt 0" ::: "memory");
#endif
}

// ---------------------------------------------------------------------------
// 1) Gather K context embeddings per (t,b), convert fp32 -> bf16 into X.
//    One block per output row; float4 reads, 4x bf16 packed writes.
// ---------------------------------------------------------------------------
__global__ __launch_bounds__(256)
void gather_embed_bf16(const int* __restrict__ tokens,
                       const float* __restrict__ emb,
                       __bf16* __restrict__ X)
{
    const int row = blockIdx.x;               // row = t*B + b
    const int t   = row >> 2;                 // B_ == 4
    const int b   = row & 3;

    int tok[KCTX];
#pragma unroll
    for (int ki = 0; ki < KCTX; ++ki) {
        const int src = t + ki - KCTX;        // padded[t+ki] = tokens[t+ki-K]
        tok[ki] = (src >= 0) ? tokens[src * B_ + b] : 0;  // pad token id 0
    }

    __bf16* xrow = X + (long long)row * K1_;
    for (int j = threadIdx.x; j < K1_ / 4; j += 256) {
        const int base = j * 4;
        const int ki   = base >> 10;          // E_ == 1024
        const int e    = base & (E_ - 1);
        const float4 f = *(const float4*)(emb + (long long)tok[ki] * E_ + e);
        v4bf o;
        o.x = (__bf16)f.x; o.y = (__bf16)f.y; o.z = (__bf16)f.z; o.w = (__bf16)f.w;
        *(v4bf*)(xrow + base) = o;
    }
}

// ---------------------------------------------------------------------------
// 2) LDS-tiled convert+transpose: W (Kd x N fp32, row-major) -> Wt (N x Kd bf16).
//    Coalesced reads along N, coalesced writes along Kd.
// ---------------------------------------------------------------------------
__global__ __launch_bounds__(256)
void convert_transpose_bf16(const float* __restrict__ W,
                            __bf16* __restrict__ Wt,
                            int Kd, int N)
{
    __shared__ float tile[32][33];
    const int n0 = blockIdx.x * 32;
    const int k0 = blockIdx.y * 32;
    const int tx = threadIdx.x & 31;
    const int ty = threadIdx.x >> 5;          // 8 rows of 32 threads

#pragma unroll
    for (int i = 0; i < 4; ++i) {
        const int k = ty + i * 8;
        tile[k][tx] = W[(long long)(k0 + k) * N + (n0 + tx)];
    }
    __syncthreads();
#pragma unroll
    for (int i = 0; i < 4; ++i) {
        const int n = ty + i * 8;
        Wt[(long long)(n0 + n) * Kd + (k0 + tx)] = (__bf16)tile[tx][n];
    }
}

// ---------------------------------------------------------------------------
// 3) bf16 WMMA GEMM with async-LDS double buffering.
//      C = A(M x Kd) * Bt(N x Kd)^T + bias, optional fused SiLU->bf16.
//    Block: 256 threads = 8 waves (2 x 4). Block tile 64(M) x 256(N).
//    Wave tile 32 x 64 -> 8 f32 accumulators; K-step 32.
//    LDS tiles padded to 40 elements/row (80 B) to spread banks.
//
//    16-bit A/B fragment layout (ISA 7.12.2):
//      lane L: row/col = L%16, half = L/16
//      VGPR 0..3 : K = half*8 + {0..7}       (16 contiguous bytes)
//      VGPR 4..7 : K = 16 + half*8 + {0..7}  (16 contiguous bytes)
//    f32 C/D layout: VGPR r -> (row = r + 8*half, col = L%16).
// ---------------------------------------------------------------------------
static constexpr int BM  = 64;
static constexpr int BN  = 256;
static constexpr int BK  = 32;
static constexpr int LDP = 40;                // padded row length (elements)

template <bool SILU_BF16_OUT>
__global__ __launch_bounds__(256)
void gemm_wmma_bf16(const __bf16* __restrict__ A,    // M x Kd
                    const __bf16* __restrict__ Bt,   // N x Kd
                    const float*  __restrict__ bias, // N
                    void* __restrict__ Cout,         // M x N
                    int N, int Kd)
{
    __shared__ alignas(16) __bf16 shA[2][BM * LDP];   //  2 x 5.0 KB
    __shared__ alignas(16) __bf16 shB[2][BN * LDP];   //  2 x 20.0 KB

    const int tid   = threadIdx.x;
    const int lane  = tid & 31;
    const int wave  = tid >> 5;           // 0..7
    const int waveM = wave & 1;           // 2 wave rows   (x32)
    const int waveN = wave >> 1;          // 4 wave cols   (x64)
    const int half  = lane >> 4;
    const int l16   = lane & 15;

    const int rowBlk = blockIdx.y * BM;
    const int colBlk = blockIdx.x * BN;

    const __bf16* pA = A  + (long long)rowBlk * Kd;
    const __bf16* pB = Bt + (long long)colBlk * Kd;

    // copy mapping: each thread moves one 16B chunk of A and four of B
    const int cr = tid >> 2;              // 0..63  (tile row)
    const int cq = (tid & 3) * 8;         // 0,8,16,24 (element offset in K row)

    auto issueStage = [&](int buf, int k0) {
        async_copy16(&shA[buf][cr * LDP + cq],
                     pA + (long long)cr * Kd + k0 + cq);
#pragma unroll
        for (int i = 0; i < 4; ++i) {
            const int r2 = cr + i * 64;
            async_copy16(&shB[buf][r2 * LDP + cq],
                         pB + (long long)r2 * Kd + k0 + cq);
        }
    };

    v8f acc[2][4] = {};

    const int nk = Kd >> 5;               // K-steps of 32
    issueStage(0, 0);

    for (int kt = 0; kt < nk; ++kt) {
        const int buf = kt & 1;
        if (kt + 1 < nk) {
            issueStage(buf ^ 1, (kt + 1) << 5);
            wait_async_le5();             // retire stage kt (5 newest pending)
        } else {
            wait_async_le0();
        }
        __syncthreads();                  // tile visible to all waves

        const __bf16* sA = shA[buf];
        const __bf16* sB = shB[buf];

        FragBF16 a[2], b[4];
#pragma unroll
        for (int mi = 0; mi < 2; ++mi) {
            const __bf16* p = sA + (waveM * 32 + mi * 16 + l16) * LDP + half * 8;
            a[mi].q[0] = *(const v4u*)(p);
            a[mi].q[1] = *(const v4u*)(p + 16);
        }
#pragma unroll
        for (int c = 0; c < 4; ++c) {
            const __bf16* p = sB + (waveN * 64 + c * 16 + l16) * LDP + half * 8;
            b[c].q[0] = *(const v4u*)(p);
            b[c].q[1] = *(const v4u*)(p + 16);
        }

#pragma unroll
        for (int mi = 0; mi < 2; ++mi)
#pragma unroll
            for (int c = 0; c < 4; ++c)
                acc[mi][c] = __builtin_amdgcn_wmma_f32_16x16x32_bf16(
                    false, a[mi].v, false, b[c].v,
                    (short)0, acc[mi][c], false, false);

        __syncthreads();                  // reads done before buffer reuse
    }

    // ---- epilogue: bias (+ SiLU -> bf16) or plain fp32 store ----
#pragma unroll
    for (int c = 0; c < 4; ++c) {
        const int   col = colBlk + waveN * 64 + c * 16 + l16;
        const float bv  = bias[col];
#pragma unroll
        for (int mi = 0; mi < 2; ++mi) {
#pragma unroll
            for (int r = 0; r < 8; ++r) {
                const int row = rowBlk + waveM * 32 + mi * 16 + r + half * 8;
                float v = acc[mi][c][r] + bv;
                if constexpr (SILU_BF16_OUT) {
                    const float s = v * (1.0f / (1.0f + __expf(-v)));  // SiLU
                    ((__bf16*)Cout)[(long long)row * N + col] = (__bf16)s;
                } else {
                    ((float*)Cout)[(long long)row * N + col] = v;
                }
            }
        }
    }
}

// ---------------------------------------------------------------------------
// Launch: gather -> convert W1 -> convert Wout -> GEMM1(SiLU) -> GEMM2
// Workspace layout (all offsets 256B-aligned):
//   X     : M x K1  bf16   25,165,824 B
//   W1t   : H x K1  bf16   12,582,912 B
//   WoutT : V x H   bf16  131,072,000 B
//   Hbuf  : M x H   bf16   16,777,216 B   (total ~186 MB)
// ---------------------------------------------------------------------------
extern "C" void kernel_launch(void* const* d_in, const int* in_sizes, int n_in,
                              void* d_out, int out_size, void* d_ws, size_t ws_size,
                              hipStream_t stream)
{
    const int*   tokens = (const int*)  d_in[0];
    const float* emb    = (const float*)d_in[1];
    const float* W1     = (const float*)d_in[2];
    const float* b1     = (const float*)d_in[3];
    const float* Wout   = (const float*)d_in[4];
    const float* bout   = (const float*)d_in[5];
    float*       out    = (float*)d_out;

    char* ws = (char*)d_ws;
    const size_t szX     = (size_t)M_ * K1_ * sizeof(__bf16);
    const size_t szW1t   = (size_t)H_ * K1_ * sizeof(__bf16);
    const size_t szWoutT = (size_t)V_ * H_  * sizeof(__bf16);

    __bf16* X     = (__bf16*)(ws);
    __bf16* W1t   = (__bf16*)(ws + szX);
    __bf16* WoutT = (__bf16*)(ws + szX + szW1t);
    __bf16* Hbuf  = (__bf16*)(ws + szX + szW1t + szWoutT);

    // 1) gather + bf16 convert of activations
    gather_embed_bf16<<<M_, 256, 0, stream>>>(tokens, emb, X);

    // 2) weight converts (LDS-tiled transpose, bf16 out)
    {
        dim3 grid(H_ / 32, K1_ / 32);
        convert_transpose_bf16<<<grid, 256, 0, stream>>>(W1, W1t, K1_, H_);
    }
    {
        dim3 grid(V_ / 32, H_ / 32);
        convert_transpose_bf16<<<grid, 256, 0, stream>>>(Wout, WoutT, H_, V_);
    }

    // 3) GEMM1: Hbuf = silu(X @ W1 + b1)   (4096 x 2048, bf16 out)
    {
        dim3 grid(H_ / BN, M_ / BM);
        gemm_wmma_bf16<true><<<grid, 256, 0, stream>>>(X, W1t, b1, Hbuf, H_, K1_);
    }

    // 4) GEMM2: out = Hbuf @ Wout + bout   (4096 x 32000, fp32 out)
    {
        dim3 grid(V_ / BN, M_ / BM);
        gemm_wmma_bf16<false><<<grid, 256, 0, stream>>>(Hbuf, WoutT, bout, out, V_, H_);
    }
}
